// LSTM_24799141167558
// MI455X (gfx1250) — matrix operand order repaired and checked by reading, not compiled, not run on hardware
//
#include <hip/hip_runtime.h>
#include <hip/hip_bf16.h>

// ---------------------------------------------------------------------------
// Problem constants (from reference)
// ---------------------------------------------------------------------------
#define BATCH   2048
#define IN_DIM  8192          // SEQ*DDIM
#define SEQ     256
#define DDIM    32
#define HID     512
#define GDIM    2048          // 4*HID
#define BN_EPS  1e-5f

#define USE_TDM 1

// ---------------------------------------------------------------------------
// Workspace layout (bytes)
// ---------------------------------------------------------------------------
#define WS_SCALE  ((size_t)0)               // 8192 f32
#define WS_SHIFT  ((size_t)32*1024)        // 8192 f32
#define WS_BIASG  ((size_t)64*1024)        // 2048 f32 (b_ih+b_hh)
#define WS_WIHF   ((size_t)128*1024)       // 128 ntiles * 32 lanes * 16 bf16 = 128KB
#define WS_WHHF   ((size_t)256*1024)       // 128 ntiles * 16 kc * 32 * 16 bf16 = 2MB
#define WS_XN     ((size_t)4*1024*1024)    // 2048*8192 bf16 = 32MB
#define WS_XS     ((size_t)40*1024*1024)   // 2048*8192 bf16 = 32MB  (end: 72MB)

typedef __attribute__((ext_vector_type(16))) __bf16        v16bf;
typedef __attribute__((ext_vector_type(8)))  float         v8f;
typedef __attribute__((ext_vector_type(4)))  unsigned int  u32x4;
typedef __attribute__((ext_vector_type(8)))  unsigned int  u32x8;

__device__ __forceinline__ unsigned short f2bf(float x) {
    unsigned int u = __float_as_uint(x);
    unsigned int r = u + 0x7fffu + ((u >> 16) & 1u);   // round-to-nearest-even
    return (unsigned short)(r >> 16);
}
__device__ __forceinline__ float bf2f(unsigned short h) {
    return __uint_as_float(((unsigned int)h) << 16);
}
__device__ __forceinline__ float sigf(float x) {
    return 1.0f / (1.0f + __expf(-x));
}
__device__ __forceinline__ float tanh_fast(float x) {
    return 2.0f / (1.0f + __expf(-2.0f * x)) - 1.0f;
}
__device__ __forceinline__ void wait_tensorcnt0() {
#if __has_builtin(__builtin_amdgcn_s_wait_tensorcnt)
    __builtin_amdgcn_s_wait_tensorcnt(0);
#else
    asm volatile("s_wait_tensorcnt 0x0" ::: "memory");
#endif
}

// ---------------------------------------------------------------------------
// 1) BatchNorm column statistics -> per-feature scale/shift
// ---------------------------------------------------------------------------
__global__ __launch_bounds__(256) void bn_stats_kernel(
        const float* __restrict__ x, const float* __restrict__ gamma,
        const float* __restrict__ beta, float* __restrict__ scale,
        float* __restrict__ shift) {
    int j = blockIdx.x * 256 + threadIdx.x;   // feature
    float s = 0.f, ss = 0.f;
    for (int r = 0; r < BATCH; ++r) {
        float v = x[(size_t)r * IN_DIM + j];
        s += v; ss += v * v;
    }
    float mean = s * (1.0f / BATCH);
    float var  = ss * (1.0f / BATCH) - mean * mean;
    float rstd = rsqrtf(var + BN_EPS);
    float sc = gamma[j] * rstd;
    scale[j] = sc;
    shift[j] = beta[j] - mean * sc;
}

// ---------------------------------------------------------------------------
// 2) Normalize + cast to bf16
// ---------------------------------------------------------------------------
__global__ __launch_bounds__(256) void norm_cast_kernel(
        const float* __restrict__ x, const float* __restrict__ scale,
        const float* __restrict__ shift, unsigned short* __restrict__ xn) {
    size_t i0 = ((size_t)blockIdx.x * 256 + threadIdx.x) * 4;
    float4 v = *(const float4*)(x + i0);
    int c0 = (int)(i0 & (IN_DIM - 1));
    unsigned short r[4];
    r[0] = f2bf(v.x * scale[c0 + 0] + shift[c0 + 0]);
    r[1] = f2bf(v.y * scale[c0 + 1] + shift[c0 + 1]);
    r[2] = f2bf(v.z * scale[c0 + 2] + shift[c0 + 2]);
    r[3] = f2bf(v.w * scale[c0 + 3] + shift[c0 + 3]);
    *(uint2*)(xn + i0) = *(uint2*)r;
}

// ---------------------------------------------------------------------------
// 3) Pack bias & weight fragments.
//    B-fragment layout (bf16 16x16x32 WMMA): for output tile n0..n0+15,
//    K-chunk kb..kb+31: lane l holds N = l%16, 16 bf16 over K (table order):
//       frag[(tileIdx*32 + lane)*16 + j] = W[n0+lane%16][kb+(j>>3)*16+(lane>>4)*8+(j&7)]
// ---------------------------------------------------------------------------
__global__ __launch_bounds__(256) void pack_bias_kernel(
        const float* __restrict__ b_ih, const float* __restrict__ b_hh,
        float* __restrict__ biasg) {
    int g = blockIdx.x * 256 + threadIdx.x;
    if (g < GDIM) biasg[g] = b_ih[g] + b_hh[g];
}

__global__ __launch_bounds__(256) void pack_wih_kernel(
        const float* __restrict__ W_ih, unsigned short* __restrict__ wihf) {
    int job = blockIdx.x * 256 + threadIdx.x;      // 128 ntiles * 32 lanes
    int lane = job & 31;
    int nt = job >> 5;
    unsigned short* dst = wihf + (size_t)job * 16;
    for (int j = 0; j < 16; ++j)
        dst[j] = f2bf(W_ih[(size_t)(nt * 16 + (lane & 15)) * DDIM + (j >> 3) * 16 + (lane >> 4) * 8 + (j & 7)]);
}

__global__ __launch_bounds__(256) void pack_whh_kernel(
        const float* __restrict__ W_hh, unsigned short* __restrict__ whhf) {
    int job = blockIdx.x * 256 + threadIdx.x;      // 128 nt * 16 kc * 32 lanes
    int lane = job & 31;
    int ntkc = job >> 5;
    int kc = ntkc & 15;
    int nt = ntkc >> 4;
    unsigned short* dst = whhf + (size_t)job * 16;
    for (int j = 0; j < 16; ++j)
        dst[j] = f2bf(W_hh[(size_t)(nt * 16 + (lane & 15)) * HID + kc * 32 + (j >> 3) * 16 + (lane >> 4) * 8 + (j & 7)]);
}

// ---------------------------------------------------------------------------
// 4) layer_in GEMM: XS[b,n] = bf16( sum_k xn[b,k]*W_in[n,k] + b_in[n] )
//    M=2048 N=8192 K=8192; WG tile 32x128, 8 waves, wave w -> col tile w.
// ---------------------------------------------------------------------------
__global__ __launch_bounds__(256) void gemm1_kernel(
        const unsigned short* __restrict__ xn, const float* __restrict__ Win,
        const float* __restrict__ b_in, unsigned short* __restrict__ XS) {
    __shared__ unsigned short As[32 * 40];     // 32 rows x 32 K (pad->40)
    __shared__ unsigned short Bs[32 * 136];    // 32 K x 128 N (pad->136), transposed

    const int tid  = threadIdx.x;
    const int w    = tid >> 5;
    const int lane = tid & 31;
    const int kh   = lane >> 4;
    const int l15  = lane & 15;
    const int m0   = blockIdx.y * 32;
    const int n0   = blockIdx.x * 128;

    float bias = b_in[n0 + w * 16 + l15];
    v8f acc0, acc1;
#pragma unroll
    for (int q = 0; q < 8; ++q) { acc0[q] = bias; acc1[q] = bias; }

    const int arow = tid >> 3, agrp = tid & 7;       // A staging: 32r x 8grp x 4
    const int bn = tid >> 1, bk0 = (tid & 1) * 16;   // B staging: 128n x 2 x 16k

    for (int kb = 0; kb < IN_DIM; kb += 32) {
        uint2 aval = *(const uint2*)(xn + (size_t)(m0 + arow) * IN_DIM + kb + agrp * 4);
        const float* wp = Win + (size_t)(n0 + bn) * IN_DIM + kb + bk0;
        float bvals[16];
        *(float4*)(bvals + 0)  = *(const float4*)(wp + 0);
        *(float4*)(bvals + 4)  = *(const float4*)(wp + 4);
        *(float4*)(bvals + 8)  = *(const float4*)(wp + 8);
        *(float4*)(bvals + 12) = *(const float4*)(wp + 12);
        if (kb + 32 < IN_DIM) {   // prefetch next K chunk (global_prefetch_b8)
            __builtin_prefetch(wp + 32, 0, 0);
            __builtin_prefetch(xn + (size_t)(m0 + arow) * IN_DIM + kb + 32 + agrp * 4, 0, 0);
        }
        __syncthreads();
        *(uint2*)(As + arow * 40 + agrp * 4) = aval;
#pragma unroll
        for (int k = 0; k < 16; ++k)
            Bs[(bk0 + k) * 136 + bn] = f2bf(bvals[k]);
        __syncthreads();

        v16bf bf;
#pragma unroll
        for (int s = 0; s < 16; ++s)
            ((unsigned short*)&bf)[s] = Bs[((s >> 3) * 16 + kh * 8 + (s & 7)) * 136 + w * 16 + l15];
        v16bf a0, a1;
        ((uint4*)&a0)[0] = *(const uint4*)(As + l15 * 40 + kh * 8);
        ((uint4*)&a0)[1] = *(const uint4*)(As + l15 * 40 + 16 + kh * 8);
        ((uint4*)&a1)[0] = *(const uint4*)(As + (16 + l15) * 40 + kh * 8);
        ((uint4*)&a1)[1] = *(const uint4*)(As + (16 + l15) * 40 + 16 + kh * 8);
        acc0 = __builtin_amdgcn_wmma_f32_16x16x32_bf16(false, a0, false, bf, (short)0, acc0, false, false);
        acc1 = __builtin_amdgcn_wmma_f32_16x16x32_bf16(false, a1, false, bf, (short)0, acc1, false, false);
    }

    const int col = n0 + w * 16 + l15;
#pragma unroll
    for (int q = 0; q < 8; ++q) {
        XS[(size_t)(m0 + q + 8 * kh) * IN_DIM + col]      = f2bf(acc0[q]);
        XS[(size_t)(m0 + 16 + q + 8 * kh) * IN_DIM + col] = f2bf(acc1[q]);
    }
}

// ---------------------------------------------------------------------------
// 5) Persistent LSTM recurrence + output projection.
//    128 WGs x 256 threads; WG owns 16 batch rows; c lives in VGPRs,
//    h double-buffered bf16 in LDS. Per step: gates = xs_t@W_ih^T + h@W_hh^T
//    + (b_ih+b_hh), fused cell math, h written to the other LDS buffer.
//    xs_t tile staged by the Tensor Data Mover (wave 0 issues, waits
//    TENSORcnt, workgroup barrier releases the other waves).
// ---------------------------------------------------------------------------
__global__ __launch_bounds__(256) void lstm_kernel(
        const unsigned short* __restrict__ XS,
        const unsigned short* __restrict__ wihf,
        const unsigned short* __restrict__ whhf,
        const float* __restrict__ biasg,
        const float* __restrict__ Wout,
        float* __restrict__ out) {
    __shared__ unsigned short hA[2][16 * 520];   // 16 rows x 512 (pad->520)
    __shared__ unsigned short xsA[16 * 40];      // 16 rows x 32 (pad->40)
    __shared__ float red[16 * 8];

    const int tid  = threadIdx.x;
    const int w    = tid >> 5;
    const int lane = tid & 31;
    const int kh   = lane >> 4;
    const int l15  = lane & 15;
    const int r0   = blockIdx.x * 16;

    for (int i = tid; i < 16 * 520; i += 256) hA[0][i] = 0;

    // wave w owns hidden tiles jt = w + 8*jj ; gate gi column tile = gi*32+jt
    float bias_v[4][4];
#pragma unroll
    for (int jj = 0; jj < 4; ++jj)
#pragma unroll
        for (int gi = 0; gi < 4; ++gi)
            bias_v[jj][gi] = biasg[(gi * 32 + w + 8 * jj) * 16 + l15];

    v8f creg[4];
#pragma unroll
    for (int jj = 0; jj < 4; ++jj)
#pragma unroll
        for (int q = 0; q < 8; ++q) creg[jj][q] = 0.f;

    int p = 0;
    __syncthreads();

    for (int t = 0; t < SEQ; ++t) {
#if USE_TDM
        // ---- stage xs_t (16x32 bf16, LDS rows padded to 40) via TDM ----
        if (w == 0) {
            unsigned long long ga = (unsigned long long)(size_t)XS +
                                    ((size_t)r0 * IN_DIM + (size_t)t * DDIM) * 2ull;
            unsigned int lds = (unsigned int)(size_t)(void*)xsA;
            u32x4 g0;
            g0[0] = 1u;                                   // count=1, no gather
            g0[1] = lds;                                  // lds_addr (bytes)
            g0[2] = (unsigned int)ga;                     // global_addr[31:0]
            g0[3] = (unsigned int)((ga >> 32) & 0x01FFFFFFu) | (2u << 30); // [56:32] | type=2
            u32x8 g1;
            // data_size=2B, pad_enable, pad_interval=16 dwords (64B row),
            // pad_amount=4 dwords (16B pad -> row stride 40 elements)
            g1[0] = (1u << 16) | (1u << 20) | (3u << 22) | (3u << 25);
            g1[1] = ((unsigned)IN_DIM & 0xFFFFu) << 16;   // tensor_dim0[15:0]
            g1[2] = (((unsigned)IN_DIM >> 16) & 0xFFFFu) |
                    (((unsigned)BATCH & 0xFFFFu) << 16);  // tensor_dim0[31:16] | tensor_dim1[15:0]
            g1[3] = (((unsigned)BATCH >> 16) & 0xFFFFu) |
                    (32u << 16);                          // tensor_dim1[31:16] | tile_dim0=32
            g1[4] = 16u;                                  // tile_dim1=16, tile_dim2=0
            g1[5] = (unsigned)IN_DIM;                     // tensor_dim0_stride[31:0]
            g1[6] = 0u;                                   // stride hi, dim1_stride lo
            g1[7] = 0u;
            asm volatile("tensor_load_to_lds %0, %1" :: "s"(g0), "s"(g1) : "memory");
            wait_tensorcnt0();
        }
#else
        {   // fallback: manual staging, 2 elems per thread
            int row = tid >> 4, c = (tid & 15) * 2;
            unsigned int v = *(const unsigned int*)(XS + (size_t)(r0 + row) * IN_DIM + t * DDIM + c);
            *(unsigned int*)(xsA + row * 40 + c) = v;
        }
#endif
        if (t + 1 < SEQ) {  // prefetch next step's xs tile (global_prefetch_b8)
            int row = tid >> 4;
            __builtin_prefetch(XS + (size_t)(r0 + row) * IN_DIM + (size_t)(t + 1) * DDIM, 0, 0);
        }
        __syncthreads();

#pragma unroll
        for (int jj = 0; jj < 4; ++jj) {
            const int jt = w + 8 * jj;
            v8f acc[4];
#pragma unroll
            for (int gi = 0; gi < 4; ++gi)
#pragma unroll
                for (int q = 0; q < 8; ++q) acc[gi][q] = bias_v[jj][gi];

            // K chunk from xs_t (K=32)
            v16bf af;
            ((uint4*)&af)[0] = *(const uint4*)(xsA + l15 * 40 + kh * 8);
            ((uint4*)&af)[1] = *(const uint4*)(xsA + l15 * 40 + 16 + kh * 8);
#pragma unroll
            for (int gi = 0; gi < 4; ++gi) {
                const unsigned short* bp = wihf + ((size_t)(gi * 32 + jt) * 32 + lane) * 16;
                v16bf bf;
                ((uint4*)&bf)[0] = *(const uint4*)(bp);
                ((uint4*)&bf)[1] = *(const uint4*)(bp + 8);
                acc[gi] = __builtin_amdgcn_wmma_f32_16x16x32_bf16(false, af, false, bf, (short)0, acc[gi], false, false);
            }
            // 16 K chunks from h (K=512)
            for (int kc = 0; kc < 16; ++kc) {
                v16bf ah;
                const unsigned short* hp = hA[p] + l15 * 520 + kc * 32;
                ((uint4*)&ah)[0] = *(const uint4*)(hp + kh * 8);
                ((uint4*)&ah)[1] = *(const uint4*)(hp + 16 + kh * 8);
#pragma unroll
                for (int gi = 0; gi < 4; ++gi) {
                    const unsigned short* bp =
                        whhf + (((size_t)(gi * 32 + jt) * 16 + kc) * 32 + lane) * 16;
                    v16bf bf;
                    ((uint4*)&bf)[0] = *(const uint4*)(bp);
                    ((uint4*)&bf)[1] = *(const uint4*)(bp + 8);
                    acc[gi] = __builtin_amdgcn_wmma_f32_16x16x32_bf16(false, ah, false, bf, (short)0, acc[gi], false, false);
                }
            }
            // fused LSTM cell; C-layout: VGPR q -> row q+8*kh, col l15 of tile jt
            const int col = jt * 16 + l15;
            unsigned short* hn = hA[p ^ 1];
#pragma unroll
            for (int q = 0; q < 8; ++q) {
                float ig = sigf(acc[0][q]);
                float fg = sigf(acc[1][q]);
                float gg = tanh_fast(acc[2][q]);
                float og = sigf(acc[3][q]);
                float cn = fg * creg[jj][q] + ig * gg;
                creg[jj][q] = cn;
                hn[(q + 8 * kh) * 520 + col] = f2bf(og * tanh_fast(cn));
            }
        }
        __syncthreads();
        p ^= 1;
    }

    // out[b] = sum_j h[b,j] * W_out[j]
    if (tid < 128) {
        int row = tid >> 3, seg = tid & 7;
        const unsigned short* hp = hA[p] + row * 520 + seg * 64;
        float part = 0.f;
        for (int k = 0; k < 64; ++k) part += bf2f(hp[k]) * Wout[seg * 64 + k];
        red[row * 8 + seg] = part;
    }
    __syncthreads();
    if (tid < 16) {
        float s = 0.f;
        for (int q = 0; q < 8; ++q) s += red[tid * 8 + q];
        out[r0 + tid] = s;
    }
}

// ---------------------------------------------------------------------------
// Launch
// ---------------------------------------------------------------------------
extern "C" void kernel_launch(void* const* d_in, const int* in_sizes, int n_in,
                              void* d_out, int out_size, void* d_ws, size_t ws_size,
                              hipStream_t stream) {
    const float* x     = (const float*)d_in[0];
    const float* gamma = (const float*)d_in[1];
    const float* beta  = (const float*)d_in[2];
    const float* Win   = (const float*)d_in[3];
    const float* b_in  = (const float*)d_in[4];
    const float* W_ih  = (const float*)d_in[5];
    const float* W_hh  = (const float*)d_in[6];
    const float* b_ih  = (const float*)d_in[7];
    const float* b_hh  = (const float*)d_in[8];
    const float* Wout  = (const float*)d_in[9];
    float* out = (float*)d_out;

    char* ws = (char*)d_ws;
    float*          scale = (float*)(ws + WS_SCALE);
    float*          shift = (float*)(ws + WS_SHIFT);
    float*          biasg = (float*)(ws + WS_BIASG);
    unsigned short* wihf  = (unsigned short*)(ws + WS_WIHF);
    unsigned short* whhf  = (unsigned short*)(ws + WS_WHHF);
    unsigned short* xn    = (unsigned short*)(ws + WS_XN);
    unsigned short* XS    = (unsigned short*)(ws + WS_XS);

    bn_stats_kernel<<<IN_DIM / 256, 256, 0, stream>>>(x, gamma, beta, scale, shift);
    norm_cast_kernel<<<(size_t)BATCH * IN_DIM / 1024, 256, 0, stream>>>(x, scale, shift, xn);
    pack_bias_kernel<<<GDIM / 256, 256, 0, stream>>>(b_ih, b_hh, biasg);
    pack_wih_kernel<<<(128 * 32) / 256, 256, 0, stream>>>(W_ih, wihf);
    pack_whh_kernel<<<(128 * 16 * 32) / 256, 256, 0, stream>>>(W_hh, whhf);
    gemm1_kernel<<<dim3(IN_DIM / 128, BATCH / 32), 256, 0, stream>>>(xn, Win, b_in, XS);
    lstm_kernel<<<BATCH / 16, 256, 0, stream>>>(XS, wihf, whhf, biasg, Wout, out);
}
